// QuantLinear_49770081026763
// MI455X (gfx1250) — 8-bit matrix operand order repaired and checked by reading, compiled, not run on hardware
//
#include <hip/hip_runtime.h>
#include <hip/hip_bf16.h>
#include <stdint.h>

#define M_TOK 8192
#define K_IN  4096
#define N_OUT 4096
#define NW 127.0f
#define NA 127.0f
#define EPSQ 1e-8f

typedef int v8i __attribute__((ext_vector_type(8)));

#define WS_WSCALE 0
#define WS_XABS   16384
#define WS_WQ     65536
#define WS_XQ     (65536 + (size_t)N_OUT * K_IN)

__device__ __forceinline__ int quant_clamp(float v, float scale, float lim) {
    float q = v / scale;
    q = fminf(fmaxf(q, -lim), lim);
    return (int)rintf(q);
}

__global__ void __launch_bounds__(256)
wquant_kernel(const float* __restrict__ w, float* __restrict__ w_scale,
              int8_t* __restrict__ wq) {
    __shared__ float red[256];
    const int row = blockIdx.x;
    const int t = threadIdx.x;
    const float* wr = w + (size_t)row * K_IN;

    float amax = 0.0f;
    for (int i = t; i < K_IN; i += 256) amax = fmaxf(amax, fabsf(wr[i]));
    red[t] = amax;
    __syncthreads();
    for (int s = 128; s > 0; s >>= 1) {
        if (t < s) red[t] = fmaxf(red[t], red[t + s]);
        __syncthreads();
    }
    const float scale = fmaxf(red[0], EPSQ) / NW;
    if (t == 0) w_scale[row] = scale;

    int8_t* out = wq + (size_t)row * K_IN;
    #pragma unroll
    for (int i = 0; i < 4; ++i) {
        const int c4 = (i * 256 + t) * 4;
        float4 v = *(const float4*)(wr + c4);
        int q0 = quant_clamp(v.x, scale, NW);
        int q1 = quant_clamp(v.y, scale, NW);
        int q2 = quant_clamp(v.z, scale, NW);
        int q3 = quant_clamp(v.w, scale, NW);
        int packed = (q0 & 0xFF) | ((q1 & 0xFF) << 8) | ((q2 & 0xFF) << 16) | ((q3 & 0xFF) << 24);
        *(int*)(out + c4) = packed;
    }
}

__global__ void __launch_bounds__(256)
xabsmax_kernel(const float* __restrict__ x, unsigned int* __restrict__ absmax_bits, size_t n) {
    __shared__ float red[256];
    const int t = threadIdx.x;
    float amax = 0.0f;
    for (size_t i = (size_t)blockIdx.x * 256 + t; i < n; i += (size_t)gridDim.x * 256)
        amax = fmaxf(amax, fabsf(x[i]));
    red[t] = amax;
    __syncthreads();
    for (int s = 128; s > 0; s >>= 1) {
        if (t < s) red[t] = fmaxf(red[t], red[t + s]);
        __syncthreads();
    }
    if (t == 0) atomicMax(absmax_bits, __float_as_uint(red[0]));
}

__global__ void __launch_bounds__(256)
xquant_kernel(const float* __restrict__ x, const unsigned int* __restrict__ absmax_bits,
              int8_t* __restrict__ xq, size_t n4) {
    const float scale = fmaxf(__uint_as_float(*absmax_bits), EPSQ) / NA;
    const float4* xv = (const float4*)x;
    int* qv = (int*)xq;
    for (size_t i = (size_t)blockIdx.x * 256 + threadIdx.x; i < n4;
         i += (size_t)gridDim.x * 256) {
        float4 v = xv[i];
        int q0 = quant_clamp(v.x, scale, NA);
        int q1 = quant_clamp(v.y, scale, NA);
        int q2 = quant_clamp(v.z, scale, NA);
        int q3 = quant_clamp(v.w, scale, NA);
        qv[i] = (q0 & 0xFF) | ((q1 & 0xFF) << 8) | ((q2 & 0xFF) << 16) | ((q3 & 0xFF) << 24);
    }
}

__global__ void __launch_bounds__(256)
gemm_iu8_kernel(const int8_t* __restrict__ xq, const int8_t* __restrict__ wq,
                const float* __restrict__ w_scale,
                const unsigned int* __restrict__ absmax_bits,
                float* __restrict__ y) {
    const int lane  = threadIdx.x & 31;
    const int wave  = threadIdx.x >> 5;
    const int wm    = wave >> 2;
    const int wn    = wave & 3;
    const int m0    = blockIdx.x * 128 + wm * 64;
    const int n0    = blockIdx.y * 128 + wn * 32;
    const int lhalf = lane >> 4;
    const int lmod  = lane & 15;

    int aoff[4];
    #pragma unroll
    for (int i = 0; i < 4; ++i)
        aoff[i] = (m0 + i * 16 + lmod) * K_IN + lhalf * 16;

    int boff[2];
    #pragma unroll
    for (int j = 0; j < 2; ++j)
        boff[j] = (n0 + j * 16 + lmod) * K_IN + lhalf * 16;

    v8i acc[4][2];
    const v8i zero = {0, 0, 0, 0, 0, 0, 0, 0};
    #pragma unroll
    for (int i = 0; i < 4; ++i)
        #pragma unroll
        for (int j = 0; j < 2; ++j) acc[i][j] = zero;

    for (int k0 = 0; k0 < K_IN; k0 += 64) {
        const int8_t* abase = xq + k0;
        const int8_t* bbase = wq + k0;
        v8i a[4], b[2];
        #pragma unroll
        for (int i = 0; i < 4; ++i) {
            const int8_t* p = abase + aoff[i];
            int4 t0 = *(const int4*)(p);
            int4 t1 = *(const int4*)(p + 32);
            v8i av = {t0.x, t0.y, t0.z, t0.w, t1.x, t1.y, t1.z, t1.w};
            a[i] = av;
        }
        #pragma unroll
        for (int j = 0; j < 2; ++j) {
            const int8_t* p = bbase + boff[j];
            int4 t0 = *(const int4*)(p);
            int4 t1 = *(const int4*)(p + 32);
            v8i bv = {t0.x, t0.y, t0.z, t0.w, t1.x, t1.y, t1.z, t1.w};
            b[j] = bv;
        }
        #pragma unroll
        for (int i = 0; i < 4; ++i)
            #pragma unroll
            for (int j = 0; j < 2; ++j)
                acc[i][j] = __builtin_amdgcn_wmma_i32_16x16x64_iu8(
                     true, a[i],  true, b[j], acc[i][j],
                     false,  false);
    }

    const float xs = fmaxf(__uint_as_float(*absmax_bits), EPSQ) / NA;
    const float ws0 = w_scale[n0 + lmod] * xs;
    const float ws1 = w_scale[n0 + 16 + lmod] * xs;

    #pragma unroll
    for (int i = 0; i < 4; ++i) {
        const int rbase = m0 + i * 16 + lhalf * 8;
        #pragma unroll
        for (int r = 0; r < 8; ++r) {
            float* yr = y + (size_t)(rbase + r) * N_OUT;
            yr[n0 + lmod]      = (float)acc[i][0][r] * ws0;
            yr[n0 + 16 + lmod] = (float)acc[i][1][r] * ws1;
        }
    }
}

extern "C" void kernel_launch(void* const* d_in, const int* in_sizes, int n_in,
                              void* d_out, int out_size, void* d_ws, size_t ws_size,
                              hipStream_t stream) {
    const float* x = (const float*)d_in[0];
    const float* w = (const float*)d_in[1];
    float* y = (float*)d_out;
    char* ws = (char*)d_ws;

    float* w_scale          = (float*)(ws + WS_WSCALE);
    unsigned int* xabs_bits = (unsigned int*)(ws + WS_XABS);
    int8_t* wq              = (int8_t*)(ws + WS_WQ);
    int8_t* xq              = (int8_t*)(ws + WS_XQ);

    hipMemsetAsync(xabs_bits, 0, sizeof(unsigned int), stream);

    wquant_kernel<<<N_OUT, 256, 0, stream>>>(w, w_scale, wq);

    const size_t nx = (size_t)M_TOK * K_IN;
    xabsmax_kernel<<<2048, 256, 0, stream>>>(x, xabs_bits, nx);

    xquant_kernel<<<8192, 256, 0, stream>>>(x, xabs_bits, xq, nx / 4);

    dim3 grid(M_TOK / 128, N_OUT / 128);
    gemm_iu8_kernel<<<grid, 256, 0, stream>>>(xq, wq, w_scale, xabs_bits, y);
}
